// UltraEfficientAttention_51049981281134
// MI455X (gfx1250) — compile-verified, matrix operand order checked by reading, not run on hardware
//
#include <hip/hip_runtime.h>
#include <hip/hip_bf16.h>

// MI455X / gfx1250 sparse flash attention, bf16 WMMA, f32 accumulate.
// B=1, H=16, S=4096, DH=64.
// Main kernel computes S^T = K x Q^T so the P^T probabilities feed the PV
// WMMA B-operand directly from the C-fragment layout (no LDS re-layout),
// softmax stats are per-lane scalars, masking is a post-exp multiply, and
// the whole softmax runs in the log2 domain (log2e folded into Q prescale)
// so every exponential is a single native v_exp_f32.

typedef __attribute__((ext_vector_type(16))) __bf16 v16bf;
typedef __attribute__((ext_vector_type(8)))  __bf16 v8bf;
typedef __attribute__((ext_vector_type(8)))  float  v8f;

#define S_LEN   4096
#define DHEAD   64
#define NHEADS  16
#define CHUNK   64                 // keys per main-loop iteration
#define NCHUNK  (S_LEN / CHUNK)    // 64
#define WAVES   4
#define QROWS   (WAVES * 16)       // 64 query rows per workgroup
#define NEG_BIG (-1e9f)
// 1/sqrt(64) * log2(e): scores computed directly in log2 domain.
#define QSCALE  (0.125f * 1.4426950408889634f)

#if __has_builtin(__builtin_amdgcn_global_load_async_to_lds_b128)
#define HAVE_ASYNC 1
#endif

// ---- 16B global -> LDS copy: async DMA when available, sync fallback ----
__device__ __forceinline__ void cp16(const void* g, void* l) {
#ifdef HAVE_ASYNC
  typedef __attribute__((ext_vector_type(4))) int v4i_t;
  typedef __attribute__((address_space(1))) v4i_t* gas_p;  // v4i addrspace(1)*
  typedef __attribute__((address_space(3))) v4i_t* las_p;  // v4i addrspace(3)*
  // generic->AS1: value-identical 64-bit VA.
  // generic->AS3: ISA 10.2: low 32 bits are the LDS offset.
  gas_p gp = (gas_p)(unsigned long long)g;
  las_p lp = (las_p)(unsigned int)(unsigned long long)l;
  __builtin_amdgcn_global_load_async_to_lds_b128(gp, lp, 0, 0);
#else
  *(v8bf*)l = *(const v8bf*)g;
#endif
}

__device__ __forceinline__ void async_wait0() {
#ifdef HAVE_ASYNC
#if __has_builtin(__builtin_amdgcn_s_wait_asynccnt)
  __builtin_amdgcn_s_wait_asynccnt(0);
#else
  asm volatile("s_wait_asynccnt 0x0" ::: "memory");
#endif
#endif
}

// Native exp2: one v_exp_f32, no log2e pre-multiply.
__device__ __forceinline__ float fast_exp2(float x) {
#if __has_builtin(__builtin_amdgcn_exp2f)
  return __builtin_amdgcn_exp2f(x);
#else
  return exp2f(x);
#endif
}

// Build a 16-element bf16 fragment from two 16B LDS reads.
__device__ __forceinline__ v16bf make_frag(const __bf16* p0, const __bf16* p1) {
  v8bf a = *(const v8bf*)p0;
  v8bf b = *(const v8bf*)p1;
  v16bf r;
#pragma unroll
  for (int i = 0; i < 8; ++i) { r[i] = a[i]; r[i + 8] = b[i]; }
  return r;
}

// Convert 8 consecutive f32 (times scale) to bf16 frag elements [base, base+8).
__device__ __forceinline__ void cvt8s(const float* sp, v16bf& dst, int base, float sc) {
  float4 f0 = *(const float4*)(sp);
  float4 f1 = *(const float4*)(sp + 4);
  dst[base + 0] = (__bf16)(f0.x * sc); dst[base + 1] = (__bf16)(f0.y * sc);
  dst[base + 2] = (__bf16)(f0.z * sc); dst[base + 3] = (__bf16)(f0.w * sc);
  dst[base + 4] = (__bf16)(f1.x * sc); dst[base + 5] = (__bf16)(f1.y * sc);
  dst[base + 6] = (__bf16)(f1.z * sc); dst[base + 7] = (__bf16)(f1.w * sc);
}

// ============================== Prep kernels ==============================

// Per (head, 64-key chunk): K f32 -> bf16 straight copy; V f32 -> bf16
// transposed [feat][key] per chunk (LDS transpose, coalesced global store).
__global__ __launch_bounds__(256, 1)
void prep_kv_kernel(const float* __restrict__ K, const float* __restrict__ V,
                    __bf16* __restrict__ Kbf, __bf16* __restrict__ Vtb) {
  __shared__ __bf16 t[CHUNK * DHEAD];   // 8 KB transposed V tile
  const int h = blockIdx.x / NCHUNK;
  const int c = blockIdx.x % NCHUNK;
  const size_t base = ((size_t)h * S_LEN + (size_t)c * CHUNK) * DHEAD;
#pragma unroll
  for (int i = 0; i < 4; ++i) {
    const int u = threadIdx.x + i * 256;        // float4 unit 0..1023
    const float4 kf = *(const float4*)(K + base + (size_t)u * 4);
    __bf16* kd = Kbf + base + (size_t)u * 4;
    kd[0] = (__bf16)kf.x; kd[1] = (__bf16)kf.y;
    kd[2] = (__bf16)kf.z; kd[3] = (__bf16)kf.w;
    const float4 vf = *(const float4*)(V + base + (size_t)u * 4);
    const int key  = u >> 4;
    const int feat = (u & 15) << 2;
    t[(feat + 0) * CHUNK + key] = (__bf16)vf.x;
    t[(feat + 1) * CHUNK + key] = (__bf16)vf.y;
    t[(feat + 2) * CHUNK + key] = (__bf16)vf.z;
    t[(feat + 3) * CHUNK + key] = (__bf16)vf.w;
  }
  __syncthreads();
  __bf16* vd = Vtb + ((size_t)h * NCHUNK + c) * (CHUNK * DHEAD);
#pragma unroll
  for (int i = 0; i < 2; ++i) {
    const int u = threadIdx.x + i * 256;        // 16B unit 0..511
    *(v8bf*)(vd + u * 8) = *(const v8bf*)(&t[u * 8]);
  }
}

// CPAT[q][k] = (PAT[q][k] && MASK[k]) ? 1 : 0
__global__ __launch_bounds__(256, 1)
void prep_pat_kernel(const unsigned char* __restrict__ PAT,
                     const int* __restrict__ MASK,
                     unsigned char* __restrict__ CPAT) {
  const size_t row = blockIdx.x;
  const int k0 = threadIdx.x * 16;
  const size_t off = row * S_LEN + k0;
  union { unsigned char b[16]; uint4 u4; } pi, po;
  pi.u4 = *(const uint4*)(PAT + off);
#pragma unroll
  for (int j = 0; j < 16; ++j)
    po.b[j] = (pi.b[j] != 0 && MASK[k0 + j] != 0) ? 1 : 0;
  *(uint4*)(CPAT + off) = po.u4;
}

// ============================== Main kernel ==============================

__global__ __launch_bounds__(WAVES * 32, 1)
void attn_main_kernel(const float* __restrict__ Q,
                      const __bf16* __restrict__ Kbf,   // [H][S][DH] bf16
                      const __bf16* __restrict__ Vtb,   // [H][chunk][DH][64] bf16
                      const unsigned char* __restrict__ CPAT, // [S][S] u8
                      float* __restrict__ OUT)
{
  __shared__ __bf16 Kb[2][CHUNK * DHEAD];        // 16 KB  [key][feat]
  __shared__ __bf16 Vt[2][DHEAD * CHUNK];        // 16 KB  [feat][key]
  __shared__ unsigned char Pt[2][QROWS * CHUNK]; //  8 KB  [qrow][key]

  const int h     = blockIdx.x / (S_LEN / QROWS);
  const int qt    = blockIdx.x % (S_LEN / QROWS);
  const int qbase = qt * QROWS;
  const int tid   = threadIdx.x;
  const int wave  = tid >> 5;
  const int lane  = tid & 31;
  const int half  = lane >> 4;
  const int n     = lane & 15;    // this lane's query column (q = qbase+wave*16+n)

  // Q fragments (B-operand for S^T), prescaled by (1/sqrt(64))*log2e so all
  // downstream exponentials are native exp2.
  const float* qptr = Q + ((size_t)h * S_LEN + qbase + wave * 16 + n) * DHEAD;
  v16bf aq[2];
#pragma unroll
  for (int c = 0; c < 2; ++c) {
    cvt8s(qptr + c * 32 +      half * 8, aq[c], 0, QSCALE);
    cvt8s(qptr + c * 32 + 16 + half * 8, aq[c], 8, QSCALE);
  }

  // Per-lane scalar softmax state (one query column per lane), log2 domain.
  float run_m = -3.0e38f, rsum = 0.0f;
  v8f ot[4];                      // O^T accumulators: lane q=n, f = t*16+v+half*8
#pragma unroll
  for (int t = 0; t < 4; ++t)
#pragma unroll
    for (int v = 0; v < 8; ++v) ot[t][v] = 0.0f;

  const __bf16* kg_head = Kbf + (size_t)h * S_LEN * DHEAD;
  const __bf16* vg_head = Vtb + (size_t)h * S_LEN * DHEAD;

  auto stage = [&](int p, int c) {
    const __bf16* kg = kg_head + (size_t)c * (CHUNK * DHEAD);
    const __bf16* vg = vg_head + (size_t)c * (CHUNK * DHEAD);
#pragma unroll
    for (int i = 0; i < 4; ++i) {
      const int u = tid + i * 128;               // 16B units 0..511
      cp16(kg + u * 8, &Kb[p][u * 8]);
      cp16(vg + u * 8, &Vt[p][u * 8]);
    }
#pragma unroll
    for (int i = 0; i < 2; ++i) {
      const int u = tid + i * 128;               // 16B units 0..255
      const int row = u >> 2;
      const int seg = (u & 3) * 16;
      cp16(CPAT + (size_t)(qbase + row) * S_LEN + (size_t)c * CHUNK + seg,
           &Pt[p][row * CHUNK + seg]);
    }
  };

  stage(0, 0);

  for (int c = 0; c < NCHUNK; ++c) {
    const int p = c & 1;
    async_wait0();
    __syncthreads();                 // buffer p ready; everyone done with p^1
    if (c + 1 < NCHUNK) stage(p ^ 1, c + 1);
    if (c + 2 < NCHUNK) {            // L2 prefetch of the chunk after next
      const char* pk = (const char*)(kg_head + (size_t)(c + 2) * CHUNK * DHEAD);
      const char* pv = (const char*)(vg_head + (size_t)(c + 2) * CHUNK * DHEAD);
      __builtin_prefetch(pk + tid * 64, 0, 1);
      __builtin_prefetch(pv + tid * 64, 0, 1);
    }

    // ---- S^T = K x Q^T: 8 WMMAs. C-layout: lane holds q=n, keys v+half*8 ----
    v8f s[4];
#pragma unroll
    for (int b = 0; b < 4; ++b) {
      v8f acc;
#pragma unroll
      for (int v = 0; v < 8; ++v) acc[v] = 0.0f;
#pragma unroll
      for (int fc = 0; fc < 2; ++fc) {
        const __bf16* kp = &Kb[p][(b * 16 + n) * DHEAD + fc * 32 + half * 8];
        acc = __builtin_amdgcn_wmma_f32_16x16x32_bf16(
            false, make_frag(kp, kp + 16), false, aq[fc], (short)0, acc, false, false);
      }
      s[b] = acc;
    }

    // Pattern bytes for this lane's q: 8 consecutive per key block -> dword loads.
    unsigned pw[4][2];
#pragma unroll
    for (int b = 0; b < 4; ++b) {
      const unsigned* pp =
          (const unsigned*)&Pt[p][(wave * 16 + n) * CHUNK + b * 16 + half * 8];
      pw[b][0] = pp[0];
      pw[b][1] = pp[1];
    }

    // Raw chunk max over all 64 keys (valid upper bound; masking after exp).
    float cmax = s[0][0];
#pragma unroll
    for (int b = 0; b < 4; ++b)
#pragma unroll
      for (int v = 0; v < 8; ++v) cmax = fmaxf(cmax, s[b][v]);
    cmax = fmaxf(cmax, __shfl_xor(cmax, 16, 32));

    const float nm    = fmaxf(run_m, cmax);
    const float alpha = fast_exp2(run_m - nm);
    run_m = nm;

    // e = exp2(s - nm) * pattern01; accumulate row sum.
    float rowsum = 0.0f;
#pragma unroll
    for (int b = 0; b < 4; ++b)
#pragma unroll
      for (int v = 0; v < 8; ++v) {
        const float patf = (float)((pw[b][v >> 2] >> ((v & 3) * 8)) & 0xffu);
        const float e = fast_exp2(s[b][v] - nm) * patf;
        s[b][v] = e;
        rowsum += e;
      }
    rowsum += __shfl_xor(rowsum, 16, 32);
    rsum = rsum * alpha + rowsum;
#pragma unroll
    for (int t = 0; t < 4; ++t)
#pragma unroll
      for (int v = 0; v < 8; ++v) ot[t][v] *= alpha;

    // P^T B-fragments: exactly the S^T C-fragment data, converted to bf16.
    v16bf bp[2];
#pragma unroll
    for (int kc = 0; kc < 2; ++kc)
#pragma unroll
      for (int j = 0; j < 8; ++j) {
        bp[kc][j]     = (__bf16)s[2 * kc][j];
        bp[kc][8 + j] = (__bf16)s[2 * kc + 1][j];
      }

    // ---- O^T += V^T x P^T: 8 WMMAs (4 feature tiles x 2 key halves) ----
#pragma unroll
    for (int t = 0; t < 4; ++t)
#pragma unroll
      for (int kc = 0; kc < 2; ++kc) {
        const __bf16* vp = &Vt[p][(t * 16 + n) * CHUNK + kc * 32 + half * 8];
        ot[t] = __builtin_amdgcn_wmma_f32_16x16x32_bf16(
            false, make_frag(vp, vp + 16), false, bp[kc], (short)0, ot[t], false, false);
      }
  }

  // ---- normalize (one reciprocal) and vectorized store: 8 consecutive f per lane ----
  const float rinv = 1.0f / rsum;
  float* orow = OUT + ((size_t)h * S_LEN + qbase + wave * 16 + n) * DHEAD;
#pragma unroll
  for (int t = 0; t < 4; ++t) {
    float4 lo, hi;
    lo.x = ot[t][0] * rinv; lo.y = ot[t][1] * rinv;
    lo.z = ot[t][2] * rinv; lo.w = ot[t][3] * rinv;
    hi.x = ot[t][4] * rinv; hi.y = ot[t][5] * rinv;
    hi.z = ot[t][6] * rinv; hi.w = ot[t][7] * rinv;
    *(float4*)(orow + t * 16 + half * 8)     = lo;
    *(float4*)(orow + t * 16 + half * 8 + 4) = hi;
  }
}

// ==================== Fallback (no workspace), self-contained ====================

#define FCHUNK 32

__global__ __launch_bounds__(WAVES * 32, 1)
void attn_fallback_kernel(const float* __restrict__ Q,
                          const float* __restrict__ K,
                          const float* __restrict__ V,
                          const unsigned char* __restrict__ PAT,
                          const int* __restrict__ MASK,
                          float* __restrict__ OUT)
{
  __shared__ __bf16 Klds[FCHUNK * DHEAD];
  __shared__ __bf16 Vtlds[DHEAD * FCHUNK];
  __shared__ __bf16 Pbuf[WAVES * 16 * FCHUNK];

  const int h     = blockIdx.x / (S_LEN / QROWS);
  const int qt    = blockIdx.x % (S_LEN / QROWS);
  const int qbase = qt * QROWS;
  const int tid   = threadIdx.x;
  const int wave  = tid >> 5;
  const int lane  = tid & 31;
  const int half  = lane >> 4;
  const int n     = lane & 15;

  const float* qptr = Q + ((size_t)h * S_LEN + qbase + wave * 16 + n) * DHEAD;
  v16bf aq[2];
#pragma unroll
  for (int c = 0; c < 2; ++c) {
    cvt8s(qptr + c * 32 +      half * 8, aq[c], 0, 0.125f);
    cvt8s(qptr + c * 32 + 16 + half * 8, aq[c], 8, 0.125f);
  }

  float run_m[8], rsum[8];
  v8f o[4];
#pragma unroll
  for (int v = 0; v < 8; ++v) { run_m[v] = -3.0e38f; rsum[v] = 0.0f; }
#pragma unroll
  for (int t = 0; t < 4; ++t)
#pragma unroll
    for (int v = 0; v < 8; ++v) o[t][v] = 0.0f;

  const unsigned char* patrow =
      PAT + (size_t)(qbase + wave * 16 + half * 8) * S_LEN + n;

  for (int kb = 0; kb < S_LEN; kb += FCHUNK) {
    __syncthreads();
#pragma unroll
    for (int i = 0; i < 4; ++i) {
      const int f4   = tid + i * 128;
      const int key  = f4 >> 4;
      const int feat = (f4 & 15) << 2;
      const size_t gbase = ((size_t)h * S_LEN + kb + key) * DHEAD + feat;
      const float4 kf = *(const float4*)(K + gbase);
      __bf16* kd = &Klds[key * DHEAD + feat];
      kd[0] = (__bf16)kf.x; kd[1] = (__bf16)kf.y;
      kd[2] = (__bf16)kf.z; kd[3] = (__bf16)kf.w;
      const float4 vf = *(const float4*)(V + gbase);
      Vtlds[(feat + 0) * FCHUNK + key] = (__bf16)vf.x;
      Vtlds[(feat + 1) * FCHUNK + key] = (__bf16)vf.y;
      Vtlds[(feat + 2) * FCHUNK + key] = (__bf16)vf.z;
      Vtlds[(feat + 3) * FCHUNK + key] = (__bf16)vf.w;
    }
    __syncthreads();

    v16bf bk[2][2];
#pragma unroll
    for (int b = 0; b < 2; ++b)
#pragma unroll
      for (int c = 0; c < 2; ++c) {
        const __bf16* kp = &Klds[(b * 16 + n) * DHEAD + c * 32 + half * 8];
        bk[b][c] = make_frag(kp, kp + 16);
      }

    v8f s[2];
#pragma unroll
    for (int b = 0; b < 2; ++b) {
      v8f acc;
#pragma unroll
      for (int v = 0; v < 8; ++v) acc[v] = 0.0f;
      acc = __builtin_amdgcn_wmma_f32_16x16x32_bf16(false, aq[0], false, bk[b][0],
                                                    (short)0, acc, false, false);
      acc = __builtin_amdgcn_wmma_f32_16x16x32_bf16(false, aq[1], false, bk[b][1],
                                                    (short)0, acc, false, false);
      s[b] = acc;
    }

    const int mk0 = MASK[kb + n];
    const int mk1 = MASK[kb + n + 16];
    float cmax[8];
#pragma unroll
    for (int v = 0; v < 8; ++v) {
      const unsigned char pb0 = patrow[(size_t)v * S_LEN + kb];
      const unsigned char pb1 = patrow[(size_t)v * S_LEN + kb + 16];
      const float s0 = (pb0 && mk0) ? s[0][v] : NEG_BIG;
      const float s1 = (pb1 && mk1) ? s[1][v] : NEG_BIG;
      s[0][v] = s0; s[1][v] = s1;
      cmax[v] = fmaxf(s0, s1);
    }
#pragma unroll
    for (int off = 1; off < 16; off <<= 1)
#pragma unroll
      for (int v = 0; v < 8; ++v)
        cmax[v] = fmaxf(cmax[v], __shfl_xor(cmax[v], off, 32));

    float alphas[8], rowsum[8];
#pragma unroll
    for (int v = 0; v < 8; ++v) {
      const float nm = fmaxf(run_m[v], cmax[v]);
      alphas[v] = __expf(run_m[v] - nm);
      run_m[v]  = nm;
      const float e0 = __expf(s[0][v] - nm);
      const float e1 = __expf(s[1][v] - nm);
      s[0][v] = e0; s[1][v] = e1;
      rowsum[v] = e0 + e1;
    }
#pragma unroll
    for (int off = 1; off < 16; off <<= 1)
#pragma unroll
      for (int v = 0; v < 8; ++v)
        rowsum[v] += __shfl_xor(rowsum[v], off, 32);
#pragma unroll
    for (int v = 0; v < 8; ++v) rsum[v] = rsum[v] * alphas[v] + rowsum[v];
#pragma unroll
    for (int t = 0; t < 4; ++t)
#pragma unroll
      for (int v = 0; v < 8; ++v) o[t][v] *= alphas[v];

    {
      __bf16* pw = &Pbuf[wave * 16 * FCHUNK + (half * 8) * FCHUNK + n];
#pragma unroll
      for (int v = 0; v < 8; ++v) {
        pw[v * FCHUNK]      = (__bf16)s[0][v];
        pw[v * FCHUNK + 16] = (__bf16)s[1][v];
      }
    }
    __builtin_amdgcn_wave_barrier();
    const __bf16* pr = &Pbuf[wave * 16 * FCHUNK + n * FCHUNK + half * 8];
    const v16bf ap = make_frag(pr, pr + 16);

#pragma unroll
    for (int t = 0; t < 4; ++t) {
      const __bf16* vp = &Vtlds[(t * 16 + n) * FCHUNK + half * 8];
      o[t] = __builtin_amdgcn_wmma_f32_16x16x32_bf16(false, ap, false,
                 make_frag(vp, vp + 16), (short)0, o[t], false, false);
    }
  }

#pragma unroll
  for (int t = 0; t < 4; ++t)
#pragma unroll
    for (int v = 0; v < 8; ++v)
      OUT[((size_t)h * S_LEN + qbase + wave * 16 + v + half * 8) * DHEAD + t * 16 + n] =
          o[t][v] / rsum[v];
}

// ============================== Launcher ==============================

extern "C" void kernel_launch(void* const* d_in, const int* in_sizes, int n_in,
                              void* d_out, int out_size, void* d_ws, size_t ws_size,
                              hipStream_t stream) {
  (void)in_sizes; (void)n_in; (void)out_size;
  const float*         Q    = (const float*)d_in[0];
  const float*         K    = (const float*)d_in[1];
  const float*         V    = (const float*)d_in[2];
  const unsigned char* PAT  = (const unsigned char*)d_in[3];
  const int*           MASK = (const int*)d_in[4];
  float*               OUT  = (float*)d_out;

  const size_t KV_BYTES = (size_t)NHEADS * S_LEN * DHEAD * sizeof(__bf16); // 8 MB
  const size_t NEED = 2 * KV_BYTES + (size_t)S_LEN * S_LEN;                // 32.8 MB

  if (ws_size >= NEED) {
    __bf16* Kbf = (__bf16*)d_ws;
    __bf16* Vtb = (__bf16*)((char*)d_ws + KV_BYTES);
    unsigned char* CPAT = (unsigned char*)d_ws + 2 * KV_BYTES;
    hipLaunchKernelGGL(prep_kv_kernel, dim3(NHEADS * NCHUNK), dim3(256), 0, stream,
                       K, V, Kbf, Vtb);
    hipLaunchKernelGGL(prep_pat_kernel, dim3(S_LEN), dim3(256), 0, stream,
                       PAT, MASK, CPAT);
    hipLaunchKernelGGL(attn_main_kernel, dim3(NHEADS * (S_LEN / QROWS)),
                       dim3(WAVES * 32), 0, stream, Q, Kbf, Vtb, CPAT, OUT);
  } else {
    hipLaunchKernelGGL(attn_fallback_kernel, dim3(NHEADS * (S_LEN / QROWS)),
                       dim3(WAVES * 32), 0, stream, Q, K, V, PAT, MASK, OUT);
  }
}
